// DynamicGraphIpaFrameSeqMultiChiDenoiser_32152125178218
// MI455X (gfx1250) — matrix operand order repaired and checked by reading, not compiled, not run on hardware
//
#include <hip/hip_runtime.h>
#include <hip/hip_bf16.h>
#include <math.h>

// ---- problem constants (from the reference) ----
#define C_S   256
#define C_Zc  128
#define C_Hd  16
#define Hn    16
#define PQKn  8
#define PVn   12
#define NAAn  20
#define Nn    4096
#define KNNn  20
#define LRKn  40
#define E1n   (Nn * KNNn)   // 81920
#define E2n   (Nn * LRKn)   // 163840
#define EINn  (2 * C_S + C_Zc + 4)  // 644

typedef __attribute__((ext_vector_type(16))) _Float16 v16h;
typedef __attribute__((ext_vector_type(8)))  _Float16 v8h;
typedef __attribute__((ext_vector_type(8)))  float    v8f;

// A "segment" of the (virtually concatenated, optionally row-gathered) GEMM input.
struct Seg {
    const float* base;    // source matrix
    const int*   gather;  // optional row index array (nullptr -> identity)
    int ld;               // row stride of base
    int c0, c1;           // column range [c0,c1) of the virtual input this segment covers
};

// ---------------------------------------------------------------------------
// Tiled WMMA GEMM:  Y[M x Nout] = act( concat-gather(A)[M x K] @ W[K x Nout] + bias )
// Block tile 128x64, 256 threads = 8 wave32s, each wave a 32x32 macro-tile
// (2x2 of v_wmma_f32_16x16x32_f16).
// ASSUMES M % 128 == 0 (true for all call sites: 4096 / 81920 / 163840) and
// all segment c0 multiples of 4 (16B chunk alignment). K / Nout tails and
// boundary-straddling chunks fall back to index-clamped unconditional loads
// with value selects (no per-element exec masking).
// ---------------------------------------------------------------------------
#define BM 128
#define BN 64
#define BK 32

template<int NSEG, bool EXT>
__global__ __launch_bounds__(256)
void gemm_wmma_kernel(Seg s0, Seg s1, Seg s2, const float* __restrict__ extra,
                      const float* __restrict__ W, const float* __restrict__ bias,
                      float* __restrict__ Y, int M, int K, int Nout, int act)
{
    __shared__ alignas(16) _Float16 As[BM][BK + 8];   // +8 halves pad (16B) -> bank-spread
    __shared__ alignas(16) _Float16 Bt[BN][BK + 8];   // B stored transposed: Bt[n][k]

    const int t    = threadIdx.x;
    const int lane = t & 31;
    const int wave = t >> 5;
    const int wm   = wave & 3;   // 4 waves along M
    const int wn   = wave >> 2;  // 2 waves along N
    const int m0   = blockIdx.y * BM;
    const int n0   = blockIdx.x * BN;

    // ---- per-thread A staging assignment: 16-wide chunk of one row, fixed over K ----
    const int arow = t >> 1;          // 0..127
    const int acol = (t & 1) << 4;    // 0 or 16
    const int mg   = m0 + arow;       // always < M (M % 128 == 0)
    const float* pA0 = s0.base + (size_t)(s0.gather ? s0.gather[mg] : mg) * s0.ld - s0.c0;
    const float* pA1 = (NSEG > 1)
        ? s1.base + (size_t)(s1.gather ? s1.gather[mg] : mg) * s1.ld - s1.c0 : nullptr;
    const float* pA2 = (NSEG > 2)
        ? s2.base + (size_t)(s2.gather ? s2.gather[mg] : mg) * s2.ld - s2.c0 : nullptr;
    const float* pE  = EXT ? extra + (size_t)mg * K : nullptr;

    // ---- per-thread B staging assignment: 8-wide chunk of one W row per tile ----
    const int bk = t >> 3;             // 0..31 (K row within tile)
    const int bc = (t & 7) << 3;       // 0..56 (col chunk)

    v8f acc[2][2] = {};

    for (int k0 = 0; k0 < K; k0 += BK) {
        // ---- stage A tile (128x32): fused gather+concat+convert ----
        const int kb = k0 + acol;
        float vbuf[16];
        // chunk is "uniform" if no segment boundary falls strictly inside it
        bool cross = false;
        if (NSEG > 1) cross = (kb < s0.c1) && (kb + 16 > s0.c1);
        if (NSEG > 2) cross = cross || ((kb < s1.c1) && (kb + 16 > s1.c1));
        if (!cross && (kb + 16 <= K)) {
            // fast path: whole 16-chunk in one segment, 16B-aligned -> 4x b128 loads
            const float* p = pA0;
            if (NSEG > 1 && kb >= s0.c1) p = pA1;
            if (NSEG > 2 && kb >= s1.c1) p = pA2;
            const float4* pv = (const float4*)(p + kb);
            #pragma unroll
            for (int j = 0; j < 4; ++j) {
                float4 v4 = pv[j];
                vbuf[j*4+0] = v4.x; vbuf[j*4+1] = v4.y;
                vbuf[j*4+2] = v4.z; vbuf[j*4+3] = v4.w;
            }
            if (EXT) {
                const float4* ev = (const float4*)(pE + kb);
                #pragma unroll
                for (int j = 0; j < 4; ++j) {
                    float4 e4 = ev[j];
                    vbuf[j*4+0] += e4.x; vbuf[j*4+1] += e4.y;
                    vbuf[j*4+2] += e4.z; vbuf[j*4+3] += e4.w;
                }
            }
        } else {
            #pragma unroll
            for (int i = 0; i < 16; ++i) {
                int k  = kb + i;
                int kc = (k < K) ? k : (K - 1);          // clamped: load always valid
                const float* p = pA0;
                if (NSEG > 1 && kc >= s0.c1) p = pA1;
                if (NSEG > 2 && kc >= s1.c1) p = pA2;
                float v = p[kc];
                if (EXT) v += pE[kc];
                vbuf[i] = (k < K) ? v : 0.f;             // value select, no exec mask
            }
        }
        v8h h0{}, h1{};
        #pragma unroll
        for (int j = 0; j < 8; ++j) { h0[j] = (_Float16)vbuf[j]; h1[j] = (_Float16)vbuf[8 + j]; }
        *(v8h*)&As[arow][acol]     = h0;                 // 2x ds_store_b128
        *(v8h*)&As[arow][acol + 8] = h1;

        // ---- stage B tile (32x64) transposed into LDS ----
        {
            const int kg  = k0 + bk;
            const int kgc = (kg < K) ? kg : (K - 1);
            const float* wr = W + (size_t)kgc * Nout;
            float wb[8];
            if (kg < K && (n0 + bc + 8 <= Nout)) {
                const float4* pv = (const float4*)(wr + n0 + bc);
                float4 a = pv[0], b = pv[1];
                wb[0]=a.x; wb[1]=a.y; wb[2]=a.z; wb[3]=a.w;
                wb[4]=b.x; wb[5]=b.y; wb[6]=b.z; wb[7]=b.w;
            } else {
                #pragma unroll
                for (int i = 0; i < 8; ++i) {
                    int n  = n0 + bc + i;
                    int nc = (n < Nout) ? n : (Nout - 1);
                    float v = wr[nc];
                    wb[i] = (kg < K && n < Nout) ? v : 0.f;
                }
            }
            #pragma unroll
            for (int i = 0; i < 8; ++i)
                Bt[bc + i][bk] = (_Float16)wb[i];
        }
        if (k0 + BK < K)
            __builtin_prefetch(&W[(size_t)(k0 + BK + bk) * Nout + n0 + bc], 0, 1);
        __syncthreads();

        const int g  = lane >> 4;    // half-wave group (K sub-block select)
        const int lr = lane & 15;
        #pragma unroll
        for (int tm = 0; tm < 2; ++tm) {
            // A 16x32 f16 fragment: lane row M=lr, elems 0..7 -> K=g*8+0..7,
            // elems 8..15 -> K=16+g*8+0..7 (ISA 7.12.2 table)
            int mrow = wm * 32 + tm * 16 + lr;
            v8h a0 = *(const v8h*)&As[mrow][g * 8];
            v8h a1 = *(const v8h*)&As[mrow][16 + g * 8];
            v16h a = __builtin_shufflevector(a0, a1, 0,1,2,3,4,5,6,7,8,9,10,11,12,13,14,15);
            #pragma unroll
            for (int tn = 0; tn < 2; ++tn) {
                // B 32x16 f16 fragment: lane col N=lr, elems 0..7 -> K=g*8+0..7,
                // elems 8..15 -> K=16+g*8+0..7 (same as A)
                int ncol = wn * 32 + tn * 16 + lr;
                v8h b0 = *(const v8h*)&Bt[ncol][g * 8];
                v8h b1 = *(const v8h*)&Bt[ncol][16 + g * 8];
                v16h b = __builtin_shufflevector(b0, b1, 0,1,2,3,4,5,6,7,8,9,10,11,12,13,14,15);
                acc[tm][tn] = __builtin_amdgcn_wmma_f32_16x16x32_f16(
                    false, a, false, b, (short)0, acc[tm][tn], false, false);
            }
        }
        __syncthreads();
    }

    // ---- epilogue: bias + activation, C/D layout: vgpr r -> M=r+8*(lane/16), N=lane%16
    const int g  = lane >> 4;
    const int lr = lane & 15;
    #pragma unroll
    for (int tm = 0; tm < 2; ++tm) {
        #pragma unroll
        for (int tn = 0; tn < 2; ++tn) {
            int col = n0 + wn * 32 + tn * 16 + lr;
            if (col >= Nout) continue;
            float bv = bias ? bias[col] : 0.f;
            #pragma unroll
            for (int r = 0; r < 8; ++r) {
                int row = m0 + wm * 32 + tm * 16 + r + g * 8;   // always < M
                float y = acc[tm][tn][r] + bv;
                if (act == 1)      y = fmaxf(y, 0.f);
                else if (act == 2) y = tanhf(y) * 3.14159265358979f;
                Y[(size_t)row * Nout + col] = y;
            }
        }
    }
}

// ---------------------------------------------------------------------------
// Row-wise LayerNorm:  Y = LN( X*rowscale + R ) * g + b, then *postscale
// blockDim.x == C (power of two: 128/256), one block per row.
// ---------------------------------------------------------------------------
__global__ void ln_kernel(const float* __restrict__ X, const float* __restrict__ R,
                          const float* __restrict__ rowscale,
                          const float* __restrict__ g, const float* __restrict__ b,
                          float* __restrict__ Y, const float* __restrict__ postscale)
{
    const int C = blockDim.x, row = blockIdx.x, c = threadIdx.x;
    __shared__ float red[256];
    size_t off = (size_t)row * C + c;
    float x = X[off] * (rowscale ? rowscale[row] : 1.f) + (R ? R[off] : 0.f);
    red[c] = x; __syncthreads();
    for (int s = C >> 1; s > 0; s >>= 1) { if (c < s) red[c] += red[c + s]; __syncthreads(); }
    float mean = red[0] / C; __syncthreads();
    float d = x - mean;
    red[c] = d * d; __syncthreads();
    for (int s = C >> 1; s > 0; s >>= 1) { if (c < s) red[c] += red[c + s]; __syncthreads(); }
    float var = red[0] / C;
    float y = d * rsqrtf(var + 1e-5f) * g[c] + b[c];
    if (postscale) y *= postscale[row];
    Y[off] = y;
}

// ---------------------------------------------------------------------------
__global__ void masks_kernel(const unsigned char* res, const unsigned char* seq,
                             const unsigned char* rno, const unsigned char* sno,
                             const unsigned char* cno,
                             float* rmf, float* omr, float* seqm,
                             float* rnm, float* snm, float* cmf, int n)
{
    int i = blockIdx.x * blockDim.x + threadIdx.x;
    if (i >= n) return;
    float r = res[i] ? 1.f : 0.f;
    rmf[i] = r; omr[i] = 1.f - r;
    seqm[i] = seq[i] ? 1.f : 0.f;
    rnm[i]  = rno[i] ? 1.f : 0.f;
    snm[i]  = sno[i] ? 1.f : 0.f;
    cmf[i]  = cno[i] ? 1.f : 0.f;
}

// qp_g / kvp_g = rot @ p + trans, one block per node
__global__ void rigid_points_kernel(const float* __restrict__ rot, const float* __restrict__ trans,
                                    const float* __restrict__ QP, const float* __restrict__ KVP,
                                    float* __restrict__ QPg, float* __restrict__ KVPg)
{
    int n = blockIdx.x;
    const float* R = &rot[n * 9];
    const float* tr = &trans[n * 3];
    const int NQP = Hn * PQKn;          // 128 points
    const int NKV = Hn * (PQKn + PVn);  // 320 points
    for (int p = threadIdx.x; p < NQP + NKV; p += blockDim.x) {
        const float* src; float* dst;
        if (p < NQP) { src = &QP[(size_t)n * NQP * 3 + p * 3];  dst = &QPg[(size_t)n * NQP * 3 + p * 3]; }
        else { int q = p - NQP; src = &KVP[(size_t)n * NKV * 3 + q * 3]; dst = &KVPg[(size_t)n * NKV * 3 + q * 3]; }
        float x = src[0], y = src[1], z = src[2];
        dst[0] = R[0]*x + R[1]*y + R[2]*z + tr[0];
        dst[1] = R[3]*x + R[4]*y + R[5]*z + tr[1];
        dst[2] = R[6]*x + R[7]*y + R[8]*z + tr[2];
    }
}

// attention logits per (edge, head)
__global__ void attn_logits_kernel(const int* __restrict__ src, const int* __restrict__ dst,
                                   const float* __restrict__ Q, const float* __restrict__ KV,
                                   const float* __restrict__ QPg, const float* __restrict__ KVPg,
                                   const float* __restrict__ Bz, const float* __restrict__ hw,
                                   const float* __restrict__ maskv, float* __restrict__ logits, int E)
{
    int idx = blockIdx.x * blockDim.x + threadIdx.x;
    if (idx >= E * Hn) return;
    int e = idx >> 4, h = idx & 15;
    int sN = src[e], dN = dst[e];
    const float* q = &Q[(size_t)dN * 256 + h * 16];
    const float* k = &KV[(size_t)sN * 512 + h * 32];           // kv[...,0,:]
    float qk = 0.f;
    #pragma unroll
    for (int c = 0; c < 16; ++c) qk += q[c] * k[c];
    const float* qp = &QPg[(size_t)dN * 384 + h * 24];         // 8 pts x 3
    const float* kp = &KVPg[(size_t)sN * 960 + h * 60];        // first 8 of 20 pts
    float d2 = 0.f;
    #pragma unroll
    for (int p = 0; p < 24; ++p) { float df = qp[p] - kp[p]; d2 += df * df; }
    float x = hw[h];
    float sp = (x > 20.f) ? x : log1pf(expf(x));               // softplus
    logits[idx] = qk * 0.14433756729740643f                    // sqrt(1/(3*16))
                + 0.5773502691896258f * Bz[idx]                // sqrt(1/3)
                - 0.5f * d2 * sp * 0.09622504486493763f        // sqrt(1/108)
                + 1e9f * (maskv[sN] - 1.f);
}

// segment softmax: dst == repeat(arange(N), Eper) => contiguous per node
__global__ void seg_softmax_kernel(float* __restrict__ logits, int Eper)
{
    int n = blockIdx.x, h = threadIdx.x;
    if (h >= Hn) return;
    size_t base = (size_t)n * Eper * Hn + h;
    float m = -1e30f;
    for (int e = 0; e < Eper; ++e) m = fmaxf(m, logits[base + (size_t)e * Hn]);
    float s = 0.f;
    for (int e = 0; e < Eper; ++e) s += expf(logits[base + (size_t)e * Hn] - m);
    float inv = 1.f / (s + 1e-9f);
    for (int e = 0; e < Eper; ++e) {
        size_t p = base + (size_t)e * Hn;
        logits[p] = expf(logits[p] - m) * inv;
    }
}

// per-node aggregation -> 3072-wide IPA feature [o(256)|opt(576)|optn(192)|opair(2048)]
__global__ __launch_bounds__(256)
void ipa_aggregate_kernel(const int* __restrict__ src, const float* __restrict__ attn,
                          const float* __restrict__ KV, const float* __restrict__ KVPg,
                          const float* __restrict__ z, const float* __restrict__ rot,
                          const float* __restrict__ trans, float* __restrict__ feat, int Eper)
{
    __shared__ float sa[LRKn * Hn];
    __shared__ int   ss[LRKn];
    int n = blockIdx.x;
    for (int i = threadIdx.x; i < Eper * Hn; i += blockDim.x)
        sa[i] = attn[(size_t)n * Eper * Hn + i];
    for (int i = threadIdx.x; i < Eper; i += blockDim.x)
        ss[i] = src[n * Eper + i];
    __syncthreads();
    float* f = &feat[(size_t)n * 3072];
    for (int task = threadIdx.x; task < 2496; task += blockDim.x) {
        if (task < 256) {                           // o: (h,c)
            int h = task >> 4, c = task & 15;
            float acc = 0.f;
            for (int e = 0; e < Eper; ++e)
                acc += sa[e * Hn + h] * KV[(size_t)ss[e] * 512 + h * 32 + 16 + c];
            f[task] = acc;
        } else if (task < 448) {                    // opt + optn: (h,p)
            int l = task - 256, h = l / 12, p = l % 12;
            float o0 = 0.f, o1 = 0.f, o2 = 0.f;
            for (int e = 0; e < Eper; ++e) {
                float a = sa[e * Hn + h];
                const float* vp = &KVPg[(size_t)ss[e] * 960 + h * 60 + (8 + p) * 3];
                o0 += a * vp[0]; o1 += a * vp[1]; o2 += a * vp[2];
            }
            float q0 = o0 - trans[n * 3 + 0], q1 = o1 - trans[n * 3 + 1], q2 = o2 - trans[n * 3 + 2];
            const float* R = &rot[n * 9];           // R^T @ q
            float r0 = R[0]*q0 + R[3]*q1 + R[6]*q2;
            float r1 = R[1]*q0 + R[4]*q1 + R[7]*q2;
            float r2 = R[2]*q0 + R[5]*q1 + R[8]*q2;
            f[256 + l * 3 + 0] = r0; f[256 + l * 3 + 1] = r1; f[256 + l * 3 + 2] = r2;
            f[832 + l] = sqrtf(r0*r0 + r1*r1 + r2*r2 + 1e-8f);
        } else {                                    // opair: (h,cz)
            int l = task - 448, h = l >> 7, cz = l & 127;
            float acc = 0.f;
            for (int e = 0; e < Eper; ++e)
                acc += sa[e * Hn + h] * z[(size_t)(n * Eper + e) * C_Zc + cz];
            f[1024 + l] = acc;
        }
    }
}

// quaternion compose
__global__ void compose_kernel(const float* __restrict__ rig, const float* __restrict__ rnm,
                               const float* __restrict__ rot, const float* __restrict__ trans,
                               float* __restrict__ orot, float* __restrict__ otr, int n)
{
    int i = blockIdx.x * blockDim.x + threadIdx.x;
    if (i >= n) return;
    float m = rnm[i];
    float b = rig[i*6+0]*m, c = rig[i*6+1]*m, d = rig[i*6+2]*m;
    float tx = rig[i*6+3]*m, ty = rig[i*6+4]*m, tz = rig[i*6+5]*m;
    float a = 1.f;
    float nr = sqrtf(a*a + b*b + c*c + d*d);
    a /= nr; b /= nr; c /= nr; d /= nr;
    float R[9] = { a*a+b*b-c*c-d*d, 2*(b*c-a*d),     2*(b*d+a*c),
                   2*(b*c+a*d),     a*a-b*b+c*c-d*d, 2*(c*d-a*b),
                   2*(b*d-a*c),     2*(c*d+a*b),     a*a-b*b-c*c+d*d };
    const float* Q = &rot[i * 9];
    #pragma unroll
    for (int r = 0; r < 3; ++r)
        #pragma unroll
        for (int cc = 0; cc < 3; ++cc)
            orot[i*9 + r*3 + cc] = Q[r*3+0]*R[cc] + Q[r*3+1]*R[3+cc] + Q[r*3+2]*R[6+cc];
    otr[i*3+0] = trans[i*3+0] + Q[0]*tx + Q[1]*ty + Q[2]*tz;
    otr[i*3+1] = trans[i*3+1] + Q[3]*tx + Q[4]*ty + Q[5]*tz;
    otr[i*3+2] = trans[i*3+2] + Q[6]*tx + Q[7]*ty + Q[8]*tz;
}

__global__ void chi_update_kernel(const float* __restrict__ ang, const float* __restrict__ chis,
                                  const float* __restrict__ cmf, const float* __restrict__ seqm,
                                  float* __restrict__ out, int n)
{
    int idx = blockIdx.x * blockDim.x + threadIdx.x;
    if (idx >= n * NAAn * 4) return;
    int i = idx / (NAAn * 4);
    float a = ang[idx];
    float c = cosf(a), s = sinf(a);
    float x = chis[idx * 2], y = chis[idx * 2 + 1];
    float nx =  c * x + s * y;
    float ny = -s * x + c * y;
    float cm = cmf[i], sm = seqm[i];
    out[idx * 2]     = (nx * cm + x * (1.f - cm)) * sm;
    out[idx * 2 + 1] = (ny * cm + y * (1.f - cm)) * sm;
}

__global__ void sp_blend_kernel(const float* __restrict__ so, const float* __restrict__ mixb,
                                const float* __restrict__ sp, const float* __restrict__ snm,
                                const float* __restrict__ seqm, float* __restrict__ out, int n)
{
    int i = blockIdx.x * blockDim.x + threadIdx.x;
    if (i >= n) return;
    float mx = -1e30f;
    for (int k = 0; k < NAAn; ++k) mx = fmaxf(mx, so[i * NAAn + k]);
    float e[NAAn]; float sum = 0.f;
    for (int k = 0; k < NAAn; ++k) { e[k] = expf(so[i * NAAn + k] - mx); sum += e[k]; }
    float mix = 1.f / (1.f + expf(-mixb[i]));
    float sn = snm[i], sm = seqm[i];
    for (int k = 0; k < NAAn; ++k) {
        float su  = e[k] / sum;
        float spv = sp[i * NAAn + k];
        float spn = spv * mix + su * (1.f - mix);
        out[i * NAAn + k] = (spn * sn + spv * (1.f - sn)) * sm;
    }
}

// ---------------------------------------------------------------------------
// host-side helpers
// ---------------------------------------------------------------------------
static inline void gemm(hipStream_t st, Seg a, Seg b, Seg c, const float* extra,
                        const float* W, const float* bias, float* Y,
                        int M, int K, int Nout, int act)
{
    dim3 grid((Nout + BN - 1) / BN, (M + BM - 1) / BM);
    if (extra)
        gemm_wmma_kernel<3, true><<<grid, 256, 0, st>>>(a, b, c, extra, W, bias, Y, M, K, Nout, act);
    else if (b.c1 > b.c0 || c.c1 > c.c0)
        gemm_wmma_kernel<3, false><<<grid, 256, 0, st>>>(a, b, c, nullptr, W, bias, Y, M, K, Nout, act);
    else
        gemm_wmma_kernel<1, false><<<grid, 256, 0, st>>>(a, b, c, nullptr, W, bias, Y, M, K, Nout, act);
}
static inline void ln(hipStream_t st, const float* X, const float* R, const float* rs,
                      const float* g, const float* b, float* Y, const float* ps, int M, int C)
{
    ln_kernel<<<M, C, 0, st>>>(X, R, rs, g, b, Y, ps);
}

struct IpaP { const float *bw,*bb,*hw,*kvw,*kvb,*kvpw,*kvpb,*ow,*ob,*qw,*qb,*qpw,*qpb; };

static void run_ipa(hipStream_t st, const float* s, const float* z,
                    const int* src, const int* dst, int E, int Eper,
                    const float* maskv, const float* rot, const float* trans, const IpaP& P,
                    float* Q, float* KV, float* QP, float* KVP, float* QPg, float* KVPg,
                    float* Bz, float* lg, float* feat, float* upd)
{
    Seg nil{nullptr, nullptr, 0, 0, 0};
    Seg ss{s, nullptr, C_S, 0, C_S};
    gemm(st, ss, nil, nil, nullptr, P.qw,   P.qb,   Q,   Nn, C_S, 256, 0);
    gemm(st, ss, nil, nil, nullptr, P.kvw,  P.kvb,  KV,  Nn, C_S, 512, 0);
    gemm(st, ss, nil, nil, nullptr, P.qpw,  P.qpb,  QP,  Nn, C_S, 384, 0);
    gemm(st, ss, nil, nil, nullptr, P.kvpw, P.kvpb, KVP, Nn, C_S, 960, 0);
    rigid_points_kernel<<<Nn, 256, 0, st>>>(rot, trans, QP, KVP, QPg, KVPg);
    Seg sz{z, nullptr, C_Zc, 0, C_Zc};
    gemm(st, sz, nil, nil, nullptr, P.bw, P.bb, Bz, E, C_Zc, Hn, 0);
    attn_logits_kernel<<<(E * Hn + 255) / 256, 256, 0, st>>>(src, dst, Q, KV, QPg, KVPg,
                                                             Bz, P.hw, maskv, lg, E);
    seg_softmax_kernel<<<Nn, 32, 0, st>>>(lg, Eper);
    ipa_aggregate_kernel<<<Nn, 256, 0, st>>>(src, lg, KV, KVPg, z, rot, trans, feat, Eper);
    Seg sf{feat, nullptr, 3072, 0, 3072};
    gemm(st, sf, nil, nil, nullptr, P.ow, P.ob, upd, Nn, 3072, C_S, 0);
}

// ---------------------------------------------------------------------------
extern "C" void kernel_launch(void* const* d_in, const int* in_sizes, int n_in,
                              void* d_out, int out_size, void* d_ws, size_t ws_size,
                              hipStream_t stream)
{
    (void)in_sizes; (void)n_in; (void)out_size; (void)ws_size;

    const float* node_features       = (const float*)d_in[0];
    const float* rot                 = (const float*)d_in[1];
    const float* trans               = (const float*)d_in[2];
    const float* seq_probs           = (const float*)d_in[3];
    const float* chis                = (const float*)d_in[4];
    const float* edge_features       = (const float*)d_in[5];
    const float* new_seq_edge_inputs = (const float*)d_in[6];
    const float* seq_edge_features   = (const float*)d_in[7];
    // 88 param leaves in jax pytree order (dict keys alphabetical) at d_in[8..95]
    auto P = [&](int i) { return (const float*)d_in[8 + i]; };
    const int* edge_index     = (const int*)d_in[96];
    const int* seq_edge_index = (const int*)d_in[97];
    const unsigned char* res_mask  = (const unsigned char*)d_in[98];
    const unsigned char* seq_mask  = (const unsigned char*)d_in[99];
    const unsigned char* res_noise = (const unsigned char*)d_in[100];
    const unsigned char* seq_noise = (const unsigned char*)d_in[101];
    const unsigned char* chi_noise = (const unsigned char*)d_in[102];

    float* out = (float*)d_out;
    const size_t O_S     = 0;
    const size_t O_ROT   = O_S     + (size_t)Nn * C_S;
    const size_t O_TRANS = O_ROT   + (size_t)Nn * 9;
    const size_t O_SP    = O_TRANS + (size_t)Nn * 3;
    const size_t O_CHIS  = O_SP    + (size_t)Nn * NAAn;
    const size_t O_ZSP   = O_CHIS  + (size_t)Nn * NAAn * 8;
    const size_t O_ZSQ   = O_ZSP   + (size_t)E1n * C_Zc;

    const int* e0  = edge_index;
    const int* e1  = edge_index + E1n;
    const int* se0 = seq_edge_index;
    const int* se1 = seq_edge_index + E2n;

    // bump allocator over d_ws
    char* wp = (char*)d_ws;
    auto alloc = [&](size_t nfloats) {
        float* p = (float*)wp;
        wp += ((nfloats * sizeof(float) + 255) & ~(size_t)255);
        return p;
    };

    float* rmf  = alloc(Nn); float* omr = alloc(Nn); float* seqm = alloc(Nn);
    float* rnm  = alloc(Nn); float* snm = alloc(Nn); float* cmf  = alloc(Nn);
    masks_kernel<<<(Nn + 255) / 256, 256, 0, stream>>>(res_mask, seq_mask, res_noise,
                                                       seq_noise, chi_noise,
                                                       rmf, omr, seqm, rnm, snm, cmf, Nn);

    float* s  = alloc((size_t)Nn * C_S);
    float* t0 = alloc((size_t)Nn * C_S);
    float* t1 = alloc((size_t)Nn * C_S);
    Seg nil{nullptr, nullptr, 0, 0, 0};

    // ---- scn MLP + init LN:  s = LN(nf + mlp3([nf|chis|sp]) * (1-rmf)) ----
    {
        Seg a{node_features, nullptr, C_S, 0, 256};
        Seg b{chis, nullptr, NAAn * 8, 256, 416};
        Seg c{seq_probs, nullptr, NAAn, 416, 436};
        gemm(stream, a, b, c, nullptr, P(66), P(67), t0, Nn, 436, C_S, 1);
    }
    { Seg a{t0, nullptr, C_S, 0, C_S}; gemm(stream, a, nil, nil, nullptr, P(68), P(69), t1, Nn, C_S, C_S, 1); }
    { Seg a{t1, nullptr, C_S, 0, C_S}; gemm(stream, a, nil, nil, nullptr, P(70), P(71), t0, Nn, C_S, C_S, 0); }
    ln(stream, t0, node_features, omr, P(26), P(27), s, nullptr, Nn, C_S);

    // ---- edge_embed -> z_sp (straight into output slot) ----
    float* u0 = alloc((size_t)E1n * C_Zc);
    float* u1 = alloc((size_t)E1n * C_Zc);
    {
        Seg a{edge_features, nullptr, C_Zc + 4, 0, C_Zc + 4};
        Seg b{s, e0, C_S, C_Zc + 4, C_Zc + 4 + C_S};
        Seg c{s, e1, C_S, C_Zc + 4 + C_S, EINn};
        gemm(stream, a, b, c, nullptr, P(8), P(9), u0, E1n, EINn, C_Zc, 1);
    }
    { Seg a{u0, nullptr, C_Zc, 0, C_Zc}; gemm(stream, a, nil, nil, nullptr, P(10), P(11), u1, E1n, C_Zc, C_Zc, 1); }
    { Seg a{u1, nullptr, C_Zc, 0, C_Zc}; gemm(stream, a, nil, nil, nullptr, P(12), P(13), u0, E1n, C_Zc, C_Zc, 0); }
    float* z_sp = out + O_ZSP;
    ln(stream, u0, nullptr, nullptr, P(14), P(15), z_sp, nullptr, E1n, C_Zc);

    // ---- seq_edge_update -> z_sq ----
    float* v0   = alloc((size_t)E2n * C_Zc);
    float* v1   = alloc((size_t)E2n * C_Zc);
    float* z_sq = alloc((size_t)E2n * C_Zc);
    {
        Seg a{new_seq_edge_inputs, nullptr, C_Zc + 4, 0, C_Zc + 4};
        Seg b{s, se0, C_S, C_Zc + 4, C_Zc + 4 + C_S};
        Seg c{s, se1, C_S, C_Zc + 4 + C_S, EINn};
        gemm(stream, a, b, c, nullptr, P(74), P(75), v0, E2n, EINn, C_Zc, 1);
    }
    { Seg a{v0, nullptr, C_Zc, 0, C_Zc}; gemm(stream, a, nil, nil, nullptr, P(76), P(77), v1, E2n, C_Zc, C_Zc, 1); }
    { Seg a{v1, nullptr, C_Zc, 0, C_Zc}; gemm(stream, a, nil, nil, nullptr, P(78), P(79), v0, E2n, C_Zc, C_Zc, 0); }
    ln(stream, v0, seq_edge_features, nullptr, P(72), P(73), z_sq, nullptr, E2n, C_Zc);

    // ---- IPA buffers ----
    float* Qb   = alloc((size_t)Nn * 256);
    float* KVb  = alloc((size_t)Nn * 512);
    float* QPb  = alloc((size_t)Nn * 384);
    float* KVPb = alloc((size_t)Nn * 960);
    float* QPg  = alloc((size_t)Nn * 384);
    float* KVPg = alloc((size_t)Nn * 960);
    float* Bz   = alloc((size_t)E2n * Hn);
    float* lg   = alloc((size_t)E2n * Hn);
    float* feat = alloc((size_t)Nn * 3072);
    float* upd  = alloc((size_t)Nn * C_S);

    // ipa_sp leaves at 28..40, ipa_sq at 41..53 (alphabetical: b,hw,kv,kvp,o,q,qp)
    IpaP Psp{P(28),P(29),P(30),P(31),P(32),P(33),P(34),P(35),P(36),P(37),P(38),P(39),P(40)};
    IpaP Psq{P(41),P(42),P(43),P(44),P(45),P(46),P(47),P(48),P(49),P(50),P(51),P(52),P(53)};

    run_ipa(stream, s, z_sp, e0, e1, E1n, KNNn, omr, rot, trans, Psp,
            Qb, KVb, QPb, KVPb, QPg, KVPg, Bz, lg, feat, upd);
    ln(stream, upd, s, rmf, P(54), P(55), s, nullptr, Nn, C_S);   // s = LN(s + upd*rmf)

    run_ipa(stream, s, z_sq, se0, se1, E2n, LRKn, rmf, rot, trans, Psq,
            Qb, KVb, QPb, KVPb, QPg, KVPg, Bz, lg, feat, upd);
    ln(stream, upd, s, rmf, P(56), P(57), s, nullptr, Nn, C_S);

    // ---- nt transition, final s = LN(s + L2(relu(L1(relu(L0 s))))) * rmf ----
    { Seg a{s,  nullptr, C_S, 0, C_S}; gemm(stream, a, nil, nil, nullptr, P(58), P(59), t0, Nn, C_S, C_S, 1); }
    { Seg a{t0, nullptr, C_S, 0, C_S}; gemm(stream, a, nil, nil, nullptr, P(60), P(61), t1, Nn, C_S, C_S, 1); }
    { Seg a{t1, nullptr, C_S, 0, C_S}; gemm(stream, a, nil, nil, nullptr, P(62), P(63), t0, Nn, C_S, C_S, 0); }
    ln(stream, t0, s, nullptr, P(64), P(65), s, rmf, Nn, C_S);
    hipMemcpyAsync(out + O_S, s, (size_t)Nn * C_S * sizeof(float),
                   hipMemcpyDeviceToDevice, stream);

    // ---- backbone update + compose ----
    float* rig = alloc((size_t)Nn * 6);
    { Seg a{s, nullptr, C_S, 0, C_S}; gemm(stream, a, nil, nil, nullptr, P(0), P(1), rig, Nn, C_S, 6, 0); }
    compose_kernel<<<(Nn + 255) / 256, 256, 0, stream>>>(rig, rnm, rot, trans,
                                                         out + O_ROT, out + O_TRANS, Nn);

    // ---- chi angles (mlp3 + tanh*pi) + rotation update ----
    float* a0 = alloc((size_t)Nn * 80);
    float* a1 = alloc((size_t)Nn * 80);
    { Seg a{s,  nullptr, C_S, 0, C_S}; gemm(stream, a, nil, nil, nullptr, P(2), P(3), a0, Nn, C_S, 80, 1); }
    { Seg a{a0, nullptr, 80, 0, 80};   gemm(stream, a, nil, nil, nullptr, P(4), P(5), a1, Nn, 80, 80, 1); }
    { Seg a{a1, nullptr, 80, 0, 80};   gemm(stream, a, nil, nil, nullptr, P(6), P(7), a0, Nn, 80, 80, 2); }
    chi_update_kernel<<<(Nn * 80 + 255) / 256, 256, 0, stream>>>(a0, chis, cmf, seqm,
                                                                 out + O_CHIS, Nn);

    // ---- edge trunk (heaviest GEMMs): z_sq_new ----
    float* hb = alloc((size_t)Nn * C_Zc);
    { Seg a{s, nullptr, C_S, 0, C_S}; gemm(stream, a, nil, nil, nullptr, P(18), P(19), hb, Nn, C_S, C_Zc, 0); }
    float* ex0 = alloc((size_t)E2n * 384);
    float* ex1 = alloc((size_t)E2n * 384);
    Seg za{z_sq, nullptr, C_Zc, 0, C_Zc};
    Seg zb{hb, se0, C_Zc, C_Zc, 256};
    Seg zc{hb, se1, C_Zc, 256, 384};
    gemm(stream, za, zb, zc, nullptr, P(22), P(23), ex0, E2n, 384, 384, 1);
    { Seg a{ex0, nullptr, 384, 0, 384}; gemm(stream, a, nil, nil, nullptr, P(24), P(25), ex1, E2n, 384, 384, 1); }
    // et_final on (ex + ez): ez re-streamed via segments, ex1 as the extra addend
    gemm(stream, za, zb, zc, ex1, P(16), P(17), v1, E2n, 384, C_Zc, 0);
    ln(stream, v1, nullptr, nullptr, P(20), P(21), out + O_ZSQ, nullptr, E2n, C_Zc);

    // ---- sequence head ----
    { Seg a{s,  nullptr, C_S, 0, C_S}; gemm(stream, a, nil, nil, nullptr, P(84), P(85), t0, Nn, C_S, C_S, 1); }
    { Seg a{t0, nullptr, C_S, 0, C_S}; gemm(stream, a, nil, nil, nullptr, P(86), P(87), t1, Nn, C_S, C_S, 1); }
    float* so = alloc((size_t)Nn * NAAn);
    float* mb = alloc((size_t)Nn);
    { Seg a{t1, nullptr, C_S, 0, C_S}; gemm(stream, a, nil, nil, nullptr, P(82), P(83), so, Nn, C_S, NAAn, 0); }
    { Seg a{t1, nullptr, C_S, 0, C_S}; gemm(stream, a, nil, nil, nullptr, P(80), P(81), mb, Nn, C_S, 1, 0); }
    sp_blend_kernel<<<(Nn + 255) / 256, 256, 0, stream>>>(so, mb, seq_probs, snm, seqm,
                                                          out + O_SP, Nn);
}
